// RGCNLayer_19387482374794
// MI455X (gfx1250) — compile-verified, matrix operand order checked by reading, not run on hardware
//
#include <hip/hip_runtime.h>
#include <hip/hip_bf16.h>
#include <math.h>

// Problem constants (match reference)
#define N_NODES  50000
#define N_EDGES  800000
#define NUM_RELS 16
#define FEAT     256
#define OUTF     256

// 50000 = 3125 * 16 exactly; 256 = 8*32 exactly; OUT groups of 64 -> 4 groups.
#define M_TILES     3125
#define N_GROUPS    4
#define GEMM_WAVES  (M_TILES * N_GROUPS)

typedef __bf16 v16bf __attribute__((ext_vector_type(16)));
typedef __bf16 v8bf  __attribute__((ext_vector_type(8)));
typedef float  v8f   __attribute__((ext_vector_type(8)));

__device__ __forceinline__ unsigned short f32_to_bf16_rne(float f) {
    unsigned u = __float_as_uint(f);
    unsigned r = 0x7FFFu + ((u >> 16) & 1u);
    return (unsigned short)((u + r) >> 16);
}

// ---- prep: h (f32) -> bf16 ------------------------------------------------
__global__ void __launch_bounds__(256) k_cvt_h(const float* __restrict__ h,
                                               unsigned short* __restrict__ hb) {
    const int total = N_NODES * FEAT;
    for (int i = blockIdx.x * blockDim.x + threadIdx.x; i < total;
         i += gridDim.x * blockDim.x)
        hb[i] = f32_to_bf16_rne(h[i]);
}

// ---- prep: weight [R][F][O] f32 -> wtb [R][O][F] bf16 (transposed) --------
__global__ void __launch_bounds__(256) k_cvt_w(const float* __restrict__ w,
                                               unsigned short* __restrict__ wtb) {
    const int total = NUM_RELS * FEAT * OUTF;
    for (int i = blockIdx.x * blockDim.x + threadIdx.x; i < total;
         i += gridDim.x * blockDim.x) {
        int r   = i >> 16;        // FEAT*OUTF == 65536
        int rem = i & 65535;
        int f   = rem >> 8;       // OUTF == 256
        int o   = rem & 255;
        wtb[(r << 16) + (o << 8) + f] = f32_to_bf16_rne(w[i]);
    }
}

// ---- gate table: gate_sig[r*N + n] = sigmoid(h[n] . gw[r]) ----------------
__global__ void __launch_bounds__(256) k_gate(const float* __restrict__ h,
                                              const float* __restrict__ gw,
                                              float* __restrict__ gate_sig) {
    int wave = blockIdx.x * (blockDim.x >> 5) + (threadIdx.x >> 5);
    if (wave >= NUM_RELS * N_NODES) return;
    int lane = threadIdx.x & 31;
    int r = wave / N_NODES;
    int n = wave - r * N_NODES;
    const float* hr = h + (size_t)n * FEAT;
    const float* gr = gw + r * FEAT;
    float acc = 0.f;
#pragma unroll
    for (int i = 0; i < FEAT / 32; ++i) {
        int c = lane + i * 32;
        acc += hr[c] * gr[c];
    }
#pragma unroll
    for (int off = 16; off > 0; off >>= 1)
        acc += __shfl_xor(acc, off, 32);
    if (lane == 0)
        gate_sig[wave] = 1.0f / (1.0f + __expf(-acc));
}

// ---- zero / relu ----------------------------------------------------------
__global__ void __launch_bounds__(256) k_zero(float* __restrict__ p, int n) {
    for (int i = blockIdx.x * blockDim.x + threadIdx.x; i < n;
         i += gridDim.x * blockDim.x)
        p[i] = 0.0f;
}
__global__ void __launch_bounds__(32) k_zero_i32(int* __restrict__ p, int n) {
    int i = threadIdx.x;
    if (i < n) p[i] = 0;
}
__global__ void __launch_bounds__(256) k_relu(float* __restrict__ p, int n) {
    for (int i = blockIdx.x * blockDim.x + threadIdx.x; i < n;
         i += gridDim.x * blockDim.x)
        p[i] = fmaxf(p[i], 0.0f);
}

// ---- edge bucketing by relation -------------------------------------------
// count with per-block LDS counters, then tiny prefix scan, then fill.
__global__ void __launch_bounds__(256) k_count(const int* __restrict__ rel,
                                               int* __restrict__ cnt) {
    __shared__ int bc[NUM_RELS];
    if (threadIdx.x < NUM_RELS) bc[threadIdx.x] = 0;
    __syncthreads();
    for (int e = blockIdx.x * blockDim.x + threadIdx.x; e < N_EDGES;
         e += gridDim.x * blockDim.x)
        atomicAdd(&bc[rel[e]], 1);
    __syncthreads();
    if (threadIdx.x < NUM_RELS) atomicAdd(&cnt[threadIdx.x], bc[threadIdx.x]);
}

__global__ void __launch_bounds__(32) k_prefix(const int* __restrict__ cnt,
                                               int* __restrict__ offs,
                                               int* __restrict__ cursor) {
    if (threadIdx.x == 0) {
        int run = 0;
        for (int r = 0; r < NUM_RELS; ++r) {
            offs[r] = run;
            cursor[r] = run;
            run += cnt[r];
        }
        offs[NUM_RELS] = run;
    }
}

__global__ void __launch_bounds__(256) k_fill(const int* __restrict__ rel,
                                              int* __restrict__ cursor,
                                              int* __restrict__ elist) {
    for (int e = blockIdx.x * blockDim.x + threadIdx.x; e < N_EDGES;
         e += gridDim.x * blockDim.x) {
        int pos = atomicAdd(&cursor[rel[e]], 1);
        elist[pos] = e;
    }
}

// ---- WMMA GEMM: T = h_bf16 [N x K] * W_r^T stored as [O x K] --------------
// One wave -> 16 rows x 64 cols of T. 8 k-steps of 32, 4 wmma each.
// A 16x32 bf16 fragment layout (ISA 7.12.2): lanes 0-15 hold row M=lane,
// K 0..7 (v0-3) and 16..23 (v4-7); lanes 16-31 hold K 8..15 and 24..31.
// B mirrors A with N in place of M; W stored [O][K] row-major makes both
// fragment loads contiguous 16B chunks.
__global__ void __launch_bounds__(256) k_gemm(const unsigned short* __restrict__ hb,
                                              const unsigned short* __restrict__ wtb,
                                              float* __restrict__ T, int r) {
    const int lane = threadIdx.x & 31;
    int wave = blockIdx.x * (blockDim.x >> 5) + (threadIdx.x >> 5);
    if (wave >= GEMM_WAVES) return;           // wave-uniform branch, EXEC stays full
    const int mTile = wave >> 2;              // 0..3124
    const int nGrp  = wave & 3;               // 0..3 -> cols nGrp*64..+63
    const int lrow  = lane & 15;
    const int khalf = (lane >> 4) << 3;       // 0 or 8

    const unsigned short* Arow = hb + (size_t)(mTile * 16 + lrow) * FEAT;
    const unsigned short* W    = wtb + ((size_t)r << 16);   // [O][K]

    v8f acc[4] = {v8f{}, v8f{}, v8f{}, v8f{}};

    for (int k0 = 0; k0 < FEAT; k0 += 32) {
        v8bf a_lo = *(const v8bf*)(Arow + k0 + khalf);
        v8bf a_hi = *(const v8bf*)(Arow + k0 + 16 + khalf);
        v16bf a = __builtin_shufflevector(a_lo, a_hi,
                     0, 1, 2, 3, 4, 5, 6, 7, 8, 9, 10, 11, 12, 13, 14, 15);
#pragma unroll
        for (int sub = 0; sub < 4; ++sub) {
            const unsigned short* Bp =
                W + (size_t)(nGrp * 64 + sub * 16 + lrow) * FEAT + k0 + khalf;
            v8bf b_lo = *(const v8bf*)(Bp);
            v8bf b_hi = *(const v8bf*)(Bp + 16);
            v16bf b = __builtin_shufflevector(b_lo, b_hi,
                         0, 1, 2, 3, 4, 5, 6, 7, 8, 9, 10, 11, 12, 13, 14, 15);
            acc[sub] = __builtin_amdgcn_wmma_f32_16x16x32_bf16(
                false, a, false, b, (short)0, acc[sub], false, false);
        }
    }

    // C/D layout: VGPR v, lanes 0-15 -> row m+v, lanes 16-31 -> row m+8+v; col = lane&15
    const int rowBase = mTile * 16 + ((lane >> 4) << 3);
    const int colBase = nGrp * 64 + lrow;
#pragma unroll
    for (int v = 0; v < 8; ++v) {
        float* Tp = T + (size_t)(rowBase + v) * OUTF + colBase;
        Tp[0]  = acc[0][v];
        Tp[16] = acc[1][v];
        Tp[32] = acc[2][v];
        Tp[48] = acc[3][v];
    }
}

// ---- edge scatter for one relation over compacted edge list ---------------
// Fixed grid; each wave grid-strides over this relation's edges.
__global__ void __launch_bounds__(256) k_scatter(const float* __restrict__ T,
                                                 const float* __restrict__ gate_sig_r,
                                                 const float* __restrict__ norm,
                                                 const int* __restrict__ src,
                                                 const int* __restrict__ dst,
                                                 const int* __restrict__ offs,
                                                 const int* __restrict__ elist,
                                                 int r,
                                                 float* __restrict__ out) {
    const int lane   = threadIdx.x & 31;
    const int wave   = blockIdx.x * (blockDim.x >> 5) + (threadIdx.x >> 5);
    const int nWaves = gridDim.x * (blockDim.x >> 5);
    const int begin  = offs[r];     // uniform scalar loads
    const int end    = offs[r + 1];
    for (int i = begin + wave; i < end; i += nWaves) {
        int e = elist[i];
        int s = src[e], d = dst[e];
        float scale = norm[e] * gate_sig_r[s];
        const float* Ts = T + (size_t)s * OUTF;
        float* od = out + (size_t)d * OUTF;
#pragma unroll
        for (int j = 0; j < OUTF / 32; ++j) {
            int c = lane + j * 32;
            atomicAdd(&od[c], Ts[c] * scale);
        }
    }
}

extern "C" void kernel_launch(void* const* d_in, const int* in_sizes, int n_in,
                              void* d_out, int out_size, void* d_ws, size_t ws_size,
                              hipStream_t stream) {
    const float* h    = (const float*)d_in[0];   // [N, 256]
    const float* norm = (const float*)d_in[1];   // [E, 1]
    const float* w    = (const float*)d_in[2];   // [R, 256, 256]
    const float* gw   = (const float*)d_in[3];   // [R, 256, 1]
    const int*   src  = (const int*)d_in[4];
    const int*   dst  = (const int*)d_in[5];
    const int*   rel  = (const int*)d_in[6];
    float* out = (float*)d_out;                  // [N, 256]

    // workspace layout (all 256B-aligned)
    char* ws = (char*)d_ws;
    const size_t off_hb    = 0;                                         // 25,600,000 B
    const size_t off_wtb   = off_hb    + (size_t)N_NODES * FEAT * 2;    //  2,097,152 B
    const size_t off_gate  = off_wtb   + (size_t)NUM_RELS * FEAT * OUTF * 2;   // 3,200,000 B
    const size_t off_T     = off_gate  + (size_t)NUM_RELS * N_NODES * 4;       // 51,200,000 B
    const size_t off_cnt   = off_T     + (size_t)N_NODES * OUTF * 4;    // 16 ints
    const size_t off_offs  = off_cnt   + 256;                           // 17 ints
    const size_t off_cur   = off_offs  + 256;                           // 16 ints
    const size_t off_elist = off_cur   + 256;                           // 3,200,000 B
    unsigned short* hb     = (unsigned short*)(ws + off_hb);
    unsigned short* wtb    = (unsigned short*)(ws + off_wtb);
    float*          gate   = (float*)(ws + off_gate);
    float*          T      = (float*)(ws + off_T);
    int*            cnt    = (int*)(ws + off_cnt);
    int*            offs   = (int*)(ws + off_offs);
    int*            cursor = (int*)(ws + off_cur);
    int*            elist  = (int*)(ws + off_elist);

    const int nOut = N_NODES * OUTF;

    // prep: bf16 conversions, gate table, zero output + counters
    k_cvt_h  <<<2048, 256, 0, stream>>>(h, hb);
    k_cvt_w  <<<2048, 256, 0, stream>>>(w, wtb);
    k_gate   <<<(NUM_RELS * N_NODES + 7) / 8, 256, 0, stream>>>(h, gw, gate);
    k_zero   <<<2048, 256, 0, stream>>>(out, nOut);
    k_zero_i32<<<1, 32, 0, stream>>>(cnt, NUM_RELS);

    // bucket edges by relation (count -> prefix -> fill)
    k_count  <<<1024, 256, 0, stream>>>(rel, cnt);
    k_prefix <<<1, 32, 0, stream>>>(cnt, offs, cursor);
    k_fill   <<<1024, 256, 0, stream>>>(rel, cursor, elist);

    // per-relation: bf16 WMMA GEMM into T, then scatter that relation's edges
    const int gemmBlocks = (GEMM_WAVES + 7) / 8;   // 8 waves / block
    for (int r = 0; r < NUM_RELS; ++r) {
        k_gemm   <<<gemmBlocks, 256, 0, stream>>>(hb, wtb, T, r);
        k_scatter<<<1024, 256, 0, stream>>>(T, gate + (size_t)r * N_NODES,
                                            norm, src, dst, offs, elist, r, out);
    }

    k_relu<<<2048, 256, 0, stream>>>(out, nOut);
}
